// GPT_78932908966066
// MI455X (gfx1250) — matrix operand order repaired and checked by reading, not compiled, not run on hardware
//
#include <hip/hip_runtime.h>
#include <math.h>
#include <stdint.h>

// ---- model dims (fixed by reference) ----
#define LNUM   2
#define HN     16
#define DMODEL 1024
#define HDIM   64
#define FFDIM  4096
#define VOCAB  32000
#define SEQ    1024
#define BATCH  4
#define ROWS   (BATCH * SEQ)   // 4096

typedef __attribute__((ext_vector_type(16))) __bf16 v16bf;
typedef __attribute__((ext_vector_type(8)))  float  v8f;
typedef unsigned u32x4 __attribute__((ext_vector_type(4)));
typedef unsigned u32x8 __attribute__((ext_vector_type(8)));

union FragAB { v16bf v; __bf16 e[16]; unsigned u[8]; u32x4 x[2]; };

#define GF_BIAS 1
#define GF_GELU 2

// ---------------------------------------------------------------------------
// TDM: issue a 2D tensor_load_to_lds (global -> LDS DMA, TENSORcnt-tracked).
// Descriptor per CDNA5 ISA ch.8: group0 (4 SGPRs) + group1 (8 SGPRs),
// VADDR2/3 = NULL (2D). dsz: 0=1B,1=2B,2=4B elements. pad_* appends LDS
// dwords after each tile row so fragment reads are bank-conflict-free.
// ---------------------------------------------------------------------------
__device__ __forceinline__ void tdm_load_2d(
    unsigned lds_addr, const void* gptr, unsigned dsz,
    unsigned tensor_d0,                 // row length of tensor (elements)
    unsigned long long stride0,         // row stride (elements)
    unsigned tile_d0, unsigned tile_d1, // tile width / #rows (elements)
    unsigned pad_interval,              // pad after 8<<i bytes stored
    unsigned pad_amount)                // (amount+1) dwords appended
{
  const unsigned long long ga = (unsigned long long)(uintptr_t)gptr;
  u32x4 g0;
  g0[0] = 1u;                                        // count=1, user mode
  g0[1] = lds_addr;                                  // lds_addr[31:0]
  g0[2] = (unsigned)(ga & 0xFFFFFFFFu);              // global_addr[31:0]
  g0[3] = (unsigned)((ga >> 32) & 0x01FFFFFFu)       // global_addr[56:32]
        | (2u << 30);                                // type = 2 (image)
  u32x8 g1;
  g1[0] = (dsz << 16)                                // data_size
        | (1u << 20)                                 // pad_enable
        | (pad_interval << 22)
        | (pad_amount << 25);
  g1[1] = (tensor_d0 & 0xFFFFu) << 16;               // tensor_dim0[15:0]
  g1[2] = (tensor_d0 >> 16)                          // tensor_dim0[31:16]
        | ((tile_d1 & 0xFFFFu) << 16);               // tensor_dim1 = tile rows
  g1[3] = (tile_d0 & 0xFFFFu) << 16;                 // tile_dim0
  g1[4] = tile_d1 & 0xFFFFu;                         // tile_dim1 (tile_dim2=0)
  g1[5] = (unsigned)(stride0 & 0xFFFFFFFFu);         // tensor_dim0_stride lo
  g1[6] = (unsigned)((stride0 >> 32) & 0xFFFFu);     // stride hi (stride1=0)
  g1[7] = 0u;
  asm volatile("tensor_load_to_lds %0, %1" :: "s"(g0), "s"(g1) : "memory");
}

// ---------------------------------------------------------------------------
// bf16 WMMA GEMM over pre-converted operands:
//   C[M,N] = epi(A[M,K](bf16) @ Wt[N,K](bf16, pre-transposed) + bias)
// block = 256 threads (8 waves); tile 128(M) x 128(N); BK = 32.
// Both tiles are [128 rows][32 bf16] staged by TDM (data_size=2B), rows
// padded 64B -> 80B. Each fragment = two ds_load_b128; no per-step cvt.
// Wave tile 32x64 -> 8x v_wmma_f32_16x16x32_bf16 per K-step.
// ---------------------------------------------------------------------------
#define BM 128
#define BN 128
#define BK 32
#define LDT 40    // padded tile row, bf16 elements (80B, multiple of 16B)

__global__ __launch_bounds__(256) void gemm_bf16(
    const __bf16* __restrict__ A, const __bf16* __restrict__ Wt,
    const float* __restrict__ bias, float* __restrict__ C,
    int M, int N, int K, int flags)
{
  __shared__ __bf16 As[2][BM * LDT];   // [m][k] bf16, padded rows
  __shared__ __bf16 Bs[2][BN * LDT];   // [n][k] bf16, padded rows

  const int tid  = threadIdx.x;
  const int lane = tid & 31;
  const int wave = tid >> 5;
  const int wm   = (wave >> 1) * 32;   // 0,32,64,96
  const int wn   = (wave & 1)  * 64;   // 0,64
  const long long bm = (long long)blockIdx.y * BM;
  const long long bn = (long long)blockIdx.x * BN;

  const unsigned ldsAbuf[2] = {
      (unsigned)(unsigned long long)(uintptr_t)&As[0][0],
      (unsigned)(unsigned long long)(uintptr_t)&As[1][0] };
  const unsigned ldsBbuf[2] = {
      (unsigned)(unsigned long long)(uintptr_t)&Bs[0][0],
      (unsigned)(unsigned long long)(uintptr_t)&Bs[1][0] };

  const __bf16* Aptr = A  + bm * K;
  const __bf16* Wptr = Wt + bn * K;

  v8f acc[2][4];
  #pragma unroll
  for (int i = 0; i < 2; ++i)
    #pragma unroll
    for (int j = 0; j < 4; ++j)
      acc[i][j] = (v8f){0.f,0.f,0.f,0.f,0.f,0.f,0.f,0.f};

  // per-lane WMMA fragment geometry (CDNA5 16-bit layouts, wave32):
  // packed pairs live at element offsets {akb..akb+7} and {16+akb..16+akb+7}
  const int aM  = lane & 15;
  const int akb = (lane >> 4) * 8;

  const int nk = K / BK;

  auto issue = [&](int t, int b) {
    // rows of 32 bf16 (64B), pad +4dw (16B) => 80B stride (interval 8<<3=64B)
    tdm_load_2d(ldsAbuf[b], Aptr + (long long)t * BK, 1,
                (unsigned)K, (unsigned long long)K, BK, BM, 3, 3);
    tdm_load_2d(ldsBbuf[b], Wptr + (long long)t * BK, 1,
                (unsigned)K, (unsigned long long)K, BK, BN, 3, 3);
  };

  if (wave == 0) {
    issue(0, 0);
    __builtin_amdgcn_s_wait_tensorcnt(0);
  }
  __syncthreads();

  for (int t = 0; t < nk; ++t) {
    const int cur = t & 1, nxt = cur ^ 1;
    if (wave == 0 && (t + 1) < nk) issue(t + 1, nxt);   // overlap DMA w/ math

    FragAB afrag[2], bfrag[4];
    #pragma unroll
    for (int i = 0; i < 2; ++i) {
      const int m = wm + i * 16 + aM;
      afrag[i].x[0] = *(const u32x4*)&As[cur][m * LDT + akb];
      afrag[i].x[1] = *(const u32x4*)&As[cur][m * LDT + 16 + akb];
    }
    #pragma unroll
    for (int j = 0; j < 4; ++j) {
      const int n = wn + j * 16 + aM;
      bfrag[j].x[0] = *(const u32x4*)&Bs[cur][n * LDT + akb];
      bfrag[j].x[1] = *(const u32x4*)&Bs[cur][n * LDT + 16 + akb];
    }
    #pragma unroll
    for (int i = 0; i < 2; ++i)
      #pragma unroll
      for (int j = 0; j < 4; ++j)
        acc[i][j] = __builtin_amdgcn_wmma_f32_16x16x32_bf16(
            false, afrag[i].v, false, bfrag[j].v,
            (short)0, acc[i][j], false, false);

    __syncthreads();
    if (wave == 0 && (t + 1) < nk)
      __builtin_amdgcn_s_wait_tensorcnt(0);
    __syncthreads();
  }

  #pragma unroll
  for (int i = 0; i < 2; ++i) {
    #pragma unroll
    for (int j = 0; j < 4; ++j) {
      const long long ncol = bn + wn + j * 16 + (lane & 15);
      const float bval = (flags & GF_BIAS) ? bias[ncol] : 0.0f;
      #pragma unroll
      for (int r = 0; r < 8; ++r) {
        const long long mrow = bm + wm + i * 16 + r + ((lane >> 4) * 8);
        float vv = acc[i][j][r] + bval;
        if (flags & GF_GELU)
          vv = 0.5f * vv * (1.0f + erff(vv * 0.70710678118654752f));
        C[mrow * N + ncol] = vv;
      }
    }
  }
}

// ---------------------------------------------------------------------------
// fp32 -> bf16 elementwise (activations). n % 1024 == 0.
// ---------------------------------------------------------------------------
__global__ __launch_bounds__(256) void cvt_bf16(
    const float* __restrict__ in, __bf16* __restrict__ out, long long n)
{
  const long long i = ((long long)blockIdx.x * 256 + threadIdx.x) * 4;
  if (i >= n) return;
  const float4 f = *(const float4*)(in + i);
  union { __bf16 t[4]; uint2 u; } pk;
  pk.t[0] = (__bf16)f.x; pk.t[1] = (__bf16)f.y;
  pk.t[2] = (__bf16)f.z; pk.t[3] = (__bf16)f.w;
  *(uint2*)(out + i) = pk.u;
}

// ---------------------------------------------------------------------------
// fp32 [R][C] -> bf16 [C][R] tiled transpose (weights), batched over z.
// R % 32 == 0, C % 32 == 0.
// ---------------------------------------------------------------------------
__global__ __launch_bounds__(256) void cvt_transpose(
    const float* __restrict__ in, __bf16* __restrict__ out,
    int R, int C, long long inStride, long long outStride)
{
  __shared__ float tile[32][33];
  const float* src = in  + (long long)blockIdx.z * inStride;
  __bf16*     dst  = out + (long long)blockIdx.z * outStride;
  const int c0 = blockIdx.x * 32, r0 = blockIdx.y * 32;
  const int tx = threadIdx.x & 31, ty = threadIdx.x >> 5;   // ty: 0..7
  #pragma unroll
  for (int rr = ty; rr < 32; rr += 8)
    tile[rr][tx] = src[(long long)(r0 + rr) * C + c0 + tx];
  __syncthreads();
  #pragma unroll
  for (int rr = ty; rr < 32; rr += 8)
    dst[(long long)(c0 + rr) * R + r0 + tx] = (__bf16)tile[tx][rr];
}

// ---------------------------------------------------------------------------
// WMMA flash attention (unchanged from round 3).
// Block = 8 waves = 128 queries of one (b,h); wave owns 16 queries.
// K/V tiles (32 keys x 64 dims fp32) staged by TDM, rows 64dw -> 66dw.
// ---------------------------------------------------------------------------
#define AT_LD 66   // padded K/V row, dwords
#define AT_PLD 34  // P scratch row, bf16 elements

__global__ __launch_bounds__(256) void attn_wmma(
    const float* __restrict__ q, const float* __restrict__ kmat,
    const float* __restrict__ vmat, float* __restrict__ o)
{
  __shared__ float  Ks[2][32 * AT_LD];
  __shared__ float  Vs[2][32 * AT_LD];
  __shared__ __bf16 Ps[8][16 * AT_PLD];

  const int lane = threadIdx.x & 31;
  const int wave = threadIdx.x >> 5;

  const int nqb = SEQ / 128;
  const int bh  = blockIdx.x / nqb;
  const int qb  = blockIdx.x % nqb;
  const int hh  = bh % HN;
  const int bb  = bh / HN;
  const int q0  = qb * 128 + wave * 16;
  const long long rowbase = (long long)bb * SEQ * DMODEL + hh * HDIM;

  const int ntBlock = (qb * 128 + 127) / 32 + 1;
  const int ntWave  = (q0 + 15) / 32 + 1;

  const int aM  = lane & 15;
  const int akb = (lane >> 4) * 8;
  const int bN  = lane & 15;
  const int bkoff = (lane >> 4) * 16;
  const int hi8 = (lane >> 4) * 8;

  FragAB qf[2];
  #pragma unroll
  for (int c = 0; c < 2; ++c) {
    const long long qrow = rowbase + (long long)(q0 + aM) * DMODEL + c * 32;
    #pragma unroll
    for (int p = 0; p < 8; ++p) {
      const int kk = (p < 4) ? (akb + 2 * p) : (16 + akb + 2 * (p - 4));
      const float2 f = *(const float2*)(q + qrow + kk);
      qf[c].e[2 * p]     = (__bf16)(f.x * 0.125f);
      qf[c].e[2 * p + 1] = (__bf16)(f.y * 0.125f);
    }
  }

  const unsigned ldsK[2] = {
      (unsigned)(unsigned long long)(uintptr_t)&Ks[0][0],
      (unsigned)(unsigned long long)(uintptr_t)&Ks[1][0] };
  const unsigned ldsV[2] = {
      (unsigned)(unsigned long long)(uintptr_t)&Vs[0][0],
      (unsigned)(unsigned long long)(uintptr_t)&Vs[1][0] };

  auto issueKV = [&](int t, int b) {
    const float* kp = kmat + rowbase + (long long)t * 32 * DMODEL;
    const float* vp = vmat + rowbase + (long long)t * 32 * DMODEL;
    tdm_load_2d(ldsK[b], kp, 2, 64, DMODEL, 64, 32, 5, 1);
    tdm_load_2d(ldsV[b], vp, 2, 64, DMODEL, 64, 32, 5, 1);
  };

  float ms[8], ls[8];
  #pragma unroll
  for (int r = 0; r < 8; ++r) { ms[r] = -1e30f; ls[r] = 0.f; }
  v8f oacc[4];
  #pragma unroll
  for (int n = 0; n < 4; ++n)
    oacc[n] = (v8f){0.f,0.f,0.f,0.f,0.f,0.f,0.f,0.f};

  if (wave == 0) {
    issueKV(0, 0);
    __builtin_amdgcn_s_wait_tensorcnt(0);
  }
  __syncthreads();

  for (int t = 0; t < ntBlock; ++t) {
    const int cur = t & 1, nxt = cur ^ 1;
    if (wave == 0 && (t + 1) < ntBlock) issueKV(t + 1, nxt);

    if (t < ntWave) {
      v8f sc[2];
      sc[0] = (v8f){0.f,0.f,0.f,0.f,0.f,0.f,0.f,0.f};
      sc[1] = (v8f){0.f,0.f,0.f,0.f,0.f,0.f,0.f,0.f};
      #pragma unroll
      for (int j = 0; j < 2; ++j) {
        const int krow = (j * 16 + bN) * AT_LD;
        #pragma unroll
        for (int c = 0; c < 2; ++c) {
          FragAB kf;
          #pragma unroll
          for (int p = 0; p < 8; ++p) {
            const float2 f =
                *(const float2*)&Ks[cur][krow + c * 32 + ((p < 4) ? (akb + 2 * p) : (16 + akb + 2 * (p - 4)))];
            kf.e[2 * p]     = (__bf16)f.x;
            kf.e[2 * p + 1] = (__bf16)f.y;
          }
          sc[j] = __builtin_amdgcn_wmma_f32_16x16x32_bf16(
              false, qf[c].v, false, kf.v, (short)0, sc[j], false, false);
        }
      }

      const int colbase = t * 32;
      #pragma unroll
      for (int j = 0; j < 2; ++j) {
        const int key = colbase + j * 16 + (lane & 15);
        #pragma unroll
        for (int r = 0; r < 8; ++r) {
          const int qrow = q0 + r + hi8;
          if (key > qrow) sc[j][r] = -1e30f;
        }
      }
      #pragma unroll
      for (int r = 0; r < 8; ++r) {
        float v = fmaxf(sc[0][r], sc[1][r]);
        v = fmaxf(v, __shfl_xor(v, 1, 32));
        v = fmaxf(v, __shfl_xor(v, 2, 32));
        v = fmaxf(v, __shfl_xor(v, 4, 32));
        v = fmaxf(v, __shfl_xor(v, 8, 32));
        const float newm = fmaxf(ms[r], v);
        const float e0 = __expf(sc[0][r] - newm);
        const float e1 = __expf(sc[1][r] - newm);
        sc[0][r] = e0; sc[1][r] = e1;
        float rs = e0 + e1;
        rs += __shfl_xor(rs, 1, 32);
        rs += __shfl_xor(rs, 2, 32);
        rs += __shfl_xor(rs, 4, 32);
        rs += __shfl_xor(rs, 8, 32);
        const float cor = __expf(ms[r] - newm);
        ls[r] = ls[r] * cor + rs;
        ms[r] = newm;
        #pragma unroll
        for (int n = 0; n < 4; ++n) oacc[n][r] *= cor;
      }

      #pragma unroll
      for (int j = 0; j < 2; ++j) {
        const int key = j * 16 + (lane & 15);
        #pragma unroll
        for (int r = 0; r < 8; ++r)
          Ps[wave][(r + hi8) * AT_PLD + key] = (__bf16)sc[j][r];
      }
      FragAB pf;
      #pragma unroll
      for (int p = 0; p < 8; ++p) {
        const int kk = (p < 4) ? (akb + 2 * p) : (16 + akb + 2 * (p - 4));
        pf.u[p] = *(const unsigned*)&Ps[wave][aM * AT_PLD + kk];
      }

      #pragma unroll
      for (int n = 0; n < 4; ++n) {
        FragAB vf;
        #pragma unroll
        for (int p = 0; p < 8; ++p) {
          const int key = (p < 4) ? (akb + 2 * p) : (16 + akb + 2 * (p - 4));
          vf.e[2 * p]     = (__bf16)Vs[cur][key * AT_LD + n * 16 + bN];
          vf.e[2 * p + 1] = (__bf16)Vs[cur][(key + 1) * AT_LD + n * 16 + bN];
        }
        oacc[n] = __builtin_amdgcn_wmma_f32_16x16x32_bf16(
            false, pf.v, false, vf.v, (short)0, oacc[n], false, false);
      }
    }

    __syncthreads();
    if (wave == 0 && (t + 1) < ntBlock)
      __builtin_amdgcn_s_wait_tensorcnt(0);
    __syncthreads();
  }

  #pragma unroll
  for (int n = 0; n < 4; ++n) {
    const int dim = n * 16 + (lane & 15);
    #pragma unroll
    for (int r = 0; r < 8; ++r) {
      const int qrow = q0 + r + hi8;
      o[rowbase + (long long)qrow * DMODEL + dim] = oacc[n][r] / ls[r];
    }
  }
}

// ---------------------------------------------------------------------------
// Fused residual-add + LayerNorm over D=1024. One block per row.
// ---------------------------------------------------------------------------
__global__ __launch_bounds__(256) void add_ln(
    const float* __restrict__ resid, const float* __restrict__ delta,
    const float* __restrict__ gamma, const float* __restrict__ beta,
    float* __restrict__ out)
{
  __shared__ float red[256];
  __shared__ float red2[256];
  const long long row = blockIdx.x;
  const int tid = threadIdx.x;

  float xv[4];
  float s = 0.f, s2 = 0.f;
  #pragma unroll
  for (int j = 0; j < 4; ++j) {
    const int i = tid + j * 256;
    const float v = resid[row * DMODEL + i] + delta[row * DMODEL + i];
    xv[j] = v; s += v; s2 += v * v;
  }
  red[tid] = s; red2[tid] = s2;
  __syncthreads();
  for (int st = 128; st > 0; st >>= 1) {
    if (tid < st) { red[tid] += red[tid + st]; red2[tid] += red2[tid + st]; }
    __syncthreads();
  }
  const float mean = red[0]  * (1.0f / DMODEL);
  const float var  = red2[0] * (1.0f / DMODEL) - mean * mean;
  const float rstd = rsqrtf(var + 1e-5f);
  #pragma unroll
  for (int j = 0; j < 4; ++j) {
    const int i = tid + j * 256;
    out[row * DMODEL + i] = (xv[j] - mean) * rstd * gamma[i] + beta[i];
  }
}

// ---------------------------------------------------------------------------
// Token + position embedding. One block per (b,s) row.
// ---------------------------------------------------------------------------
__global__ __launch_bounds__(256) void embed_kernel(
    const int* __restrict__ x, const float* __restrict__ tok,
    const float* __restrict__ pos, float* __restrict__ h)
{
  const long long row = blockIdx.x;
  const int tid = threadIdx.x;
  const long long token = x[row];
  const long long srow  = row % SEQ;
  #pragma unroll
  for (int j = 0; j < 4; ++j) {
    const int i = tid + j * 256;
    h[row * DMODEL + i] = tok[token * DMODEL + i] + pos[srow * DMODEL + i];
  }
}

// ---------------------------------------------------------------------------
extern "C" void kernel_launch(void* const* d_in, const int* in_sizes, int n_in,
                              void* d_out, int out_size, void* d_ws, size_t ws_size,
                              hipStream_t stream)
{
  (void)in_sizes; (void)n_in; (void)out_size; (void)ws_size;

  const int*   x       = (const int*)  d_in[0];
  const float* tok_emb = (const float*)d_in[1];
  const float* pos_emb = (const float*)d_in[2];
  const float* Wq   = (const float*)d_in[3];
  const float* bq   = (const float*)d_in[4];
  const float* Wk   = (const float*)d_in[5];
  const float* bk   = (const float*)d_in[6];
  const float* Wv   = (const float*)d_in[7];
  const float* bv   = (const float*)d_in[8];
  const float* Wo   = (const float*)d_in[9];
  const float* bo   = (const float*)d_in[10];
  const float* ln1w = (const float*)d_in[11];
  const float* ln1b = (const float*)d_in[12];
  const float* ln2w = (const float*)d_in[13];
  const float* ln2b = (const float*)d_in[14];
  const float* W1   = (const float*)d_in[15];
  const float* b1   = (const float*)d_in[16];
  const float* W2   = (const float*)d_in[17];
  const float* b2   = (const float*)d_in[18];
  const float* Wout = (const float*)d_in[19];
  const float* bout = (const float*)d_in[20];

  // ---- workspace: fp32 activations, then bf16 region ----
  float* ws = (float*)d_ws;
  const long long RD = (long long)ROWS * DMODEL;   // 4M
  const long long RF = (long long)ROWS * FFDIM;    // 16M
  float* h  = ws;
  float* qb = h  + RD;
  float* kb = qb + RD;
  float* vb = kb + RD;
  float* ob = vb + RD;
  float* t1 = ob + RD;
  float* f1 = t1 + RD;       // [ROWS, FF]

  __bf16* bws = (__bf16*)(f1 + RF);
  __bf16* ab  = bws;                         // activation bf16, 16M elems
  __bf16* wp  = ab + RF;
  const long long WDD = (long long)DMODEL * DMODEL;   // 1M
  const long long WDF = (long long)DMODEL * FFDIM;    // 4M
  __bf16* wtq[LNUM]; __bf16* wtk[LNUM]; __bf16* wtv[LNUM];
  __bf16* wto[LNUM]; __bf16* wt1[LNUM]; __bf16* wt2[LNUM];
  for (int l = 0; l < LNUM; ++l) {
    wtq[l] = wp; wp += WDD;
    wtk[l] = wp; wp += WDD;
    wtv[l] = wp; wp += WDD;
    wto[l] = wp; wp += WDD;
    wt1[l] = wp; wp += WDF;
    wt2[l] = wp; wp += WDF;
  }
  __bf16* wtout = wp;   // [VOCAB][DMODEL]

  const dim3 blk(256);
  const dim3 gD(DMODEL / BN, ROWS / BM);
  const dim3 gF(FFDIM  / BN, ROWS / BM);
  const dim3 gV(VOCAB  / BN, ROWS / BM);
  const int  gA = BATCH * HN * (SEQ / 128);

  // ---- weight preconversion: fp32 [K][N]-ish -> bf16 [N][K] ----
  for (int l = 0; l < LNUM; ++l) {
    const long long wqoff = (long long)l * HN * DMODEL * HDIM;
    // QKV: per-head [K=1024][64] -> [64][1024], batched over 16 heads (z)
    cvt_transpose<<<dim3(2, 32, HN), blk, 0, stream>>>(
        Wq + wqoff, wtq[l], DMODEL, HDIM,
        (long long)DMODEL * HDIM, (long long)HDIM * DMODEL);
    cvt_transpose<<<dim3(2, 32, HN), blk, 0, stream>>>(
        Wk + wqoff, wtk[l], DMODEL, HDIM,
        (long long)DMODEL * HDIM, (long long)HDIM * DMODEL);
    cvt_transpose<<<dim3(2, 32, HN), blk, 0, stream>>>(
        Wv + wqoff, wtv[l], DMODEL, HDIM,
        (long long)DMODEL * HDIM, (long long)HDIM * DMODEL);
    cvt_transpose<<<dim3(32, 32, 1), blk, 0, stream>>>(
        Wo + l * WDD, wto[l], DMODEL, DMODEL, 0, 0);
    cvt_transpose<<<dim3(FFDIM / 32, 32, 1), blk, 0, stream>>>(
        W1 + l * WDF, wt1[l], DMODEL, FFDIM, 0, 0);
    cvt_transpose<<<dim3(32, FFDIM / 32, 1), blk, 0, stream>>>(
        W2 + l * WDF, wt2[l], FFDIM, DMODEL, 0, 0);
  }
  cvt_transpose<<<dim3(VOCAB / 32, 32, 1), blk, 0, stream>>>(
      Wout, wtout, DMODEL, VOCAB, 0, 0);

  embed_kernel<<<ROWS, blk, 0, stream>>>(x, tok_emb, pos_emb, h);

  for (int l = 0; l < LNUM; ++l) {
    // h -> bf16, QKV projections
    cvt_bf16<<<(int)(RD / 1024), blk, 0, stream>>>(h, ab, RD);
    gemm_bf16<<<gD, blk, 0, stream>>>(ab, wtq[l], bq + l * DMODEL, qb,
                                      ROWS, DMODEL, DMODEL, GF_BIAS);
    gemm_bf16<<<gD, blk, 0, stream>>>(ab, wtk[l], bk + l * DMODEL, kb,
                                      ROWS, DMODEL, DMODEL, GF_BIAS);
    gemm_bf16<<<gD, blk, 0, stream>>>(ab, wtv[l], bv + l * DMODEL, vb,
                                      ROWS, DMODEL, DMODEL, GF_BIAS);

    attn_wmma<<<gA, blk, 0, stream>>>(qb, kb, vb, ob);

    cvt_bf16<<<(int)(RD / 1024), blk, 0, stream>>>(ob, ab, RD);
    gemm_bf16<<<gD, blk, 0, stream>>>(ab, wto[l], bo + l * DMODEL, t1,
                                      ROWS, DMODEL, DMODEL, GF_BIAS);
    add_ln<<<ROWS, blk, 0, stream>>>(h, t1, ln1w + l * DMODEL, ln1b + l * DMODEL, h);

    cvt_bf16<<<(int)(RD / 1024), blk, 0, stream>>>(h, ab, RD);
    gemm_bf16<<<gF, blk, 0, stream>>>(ab, wt1[l], b1 + l * FFDIM, f1,
                                      ROWS, FFDIM, DMODEL, GF_BIAS | GF_GELU);
    cvt_bf16<<<(int)(RF / 1024), blk, 0, stream>>>(f1, ab, RF);
    gemm_bf16<<<gD, blk, 0, stream>>>(ab, wt2[l], b2 + l * DMODEL, t1,
                                      ROWS, DMODEL, FFDIM, GF_BIAS);
    add_ln<<<ROWS, blk, 0, stream>>>(h, t1, ln2w + l * DMODEL, ln2b + l * DMODEL, h);
  }

  cvt_bf16<<<(int)(RD / 1024), blk, 0, stream>>>(h, ab, RD);
  gemm_bf16<<<gV, blk, 0, stream>>>(ab, wtout, bout, (float*)d_out,
                                    ROWS, VOCAB, DMODEL, GF_BIAS);
  hipMemcpyAsync((float*)d_out + (long long)ROWS * VOCAB, h,
                 RD * sizeof(float), hipMemcpyDeviceToDevice, stream);
}
